// Qwen3ASRBackendOnnx_2611340116143
// MI455X (gfx1250) — matrix operand order repaired and checked by reading, not compiled, not run on hardware
//
#include <hip/hip_runtime.h>
#include <stdint.h>

// ---------------------------------------------------------------------------
// Types for CDNA5 WMMA
// ---------------------------------------------------------------------------
typedef __attribute__((ext_vector_type(16))) __bf16 v16bf;
typedef __attribute__((ext_vector_type(8)))  float  v8f;
typedef __attribute__((ext_vector_type(4)))  int    i4v;

#define AS1 __attribute__((address_space(1)))
#define AS3 __attribute__((address_space(3)))

#define L_LAYERS 6
#define DMODEL   1024
#define NHEAD    16
#define HDIM     64
#define FDIM     4096
#define OUTDIM   2048
#define BSZ      4
#define TSEQ     1024
#define MTOK     (BSZ * TSEQ)      // 4096 tokens
#define ATT_SCALE 0.125f           // 64^-0.5
#define LN_EPS    1e-5f

// ---------------------------------------------------------------------------
// CDNA5 async global->LDS copy (ASYNCcnt-tracked), with safe fallback
// ---------------------------------------------------------------------------
#if __has_builtin(__builtin_amdgcn_global_load_async_to_lds_b128)
#define HAVE_ASYNC 1
#else
#define HAVE_ASYNC 0
#endif

#if HAVE_ASYNC
#if __has_builtin(__builtin_amdgcn_s_wait_asynccnt)
#define WAIT_ASYNC(n) __builtin_amdgcn_s_wait_asynccnt(n)
#else
#define WAIT_ASYNC(n) asm volatile("s_wait_asynccnt %0" ::"i"(n))
#endif
#else
#define WAIT_ASYNC(n) ((void)0)
#endif

__device__ __forceinline__ void cp16_async(const unsigned short* g,
                                           unsigned short* l) {
#if HAVE_ASYNC
  __builtin_amdgcn_global_load_async_to_lds_b128(
      (AS1 i4v*)g, (AS3 i4v*)l, 0, 0);
#else
  *(uint4*)l = *(const uint4*)g;
#endif
}

// ---------------------------------------------------------------------------
// Helpers
// ---------------------------------------------------------------------------
__device__ __forceinline__ unsigned short f2bf(float f) {
  unsigned u = __float_as_uint(f);
  u += 0x7fffu + ((u >> 16) & 1u);   // round-to-nearest-even
  return (unsigned short)(u >> 16);
}

__device__ __forceinline__ float gelu_tanh(float x) {
  float x3 = x * x * x;
  return 0.5f * x * (1.f + tanhf(0.7978845608028654f * (x + 0.044715f * x3)));
}

union AFrag { v16bf v; unsigned short u[16]; };

// A-matrix (16x32 bf16) element K index for element slot `e`, half `hi`
__device__ __forceinline__ int a_kidx(int e, int hi) {
  int p = e >> 1;
  return ((p & 3) << 1) + ((p >> 2) << 4) + (hi << 3) + (e & 1);
}

// ---------------------------------------------------------------------------
// fp32 -> bf16 bulk convert (weights), 8 elements per thread
// ---------------------------------------------------------------------------
__global__ __launch_bounds__(256)
void f32_to_bf16_kernel(const float* __restrict__ src,
                        unsigned short* __restrict__ dst) {
  const size_t i = ((size_t)blockIdx.x * 256 + threadIdx.x) * 8;
  float4 a = *(const float4*)(src + i);
  float4 b = *(const float4*)(src + i + 4);
  unsigned short t[8];
  t[0] = f2bf(a.x); t[1] = f2bf(a.y); t[2] = f2bf(a.z); t[3] = f2bf(a.w);
  t[4] = f2bf(b.x); t[5] = f2bf(b.y); t[6] = f2bf(b.z); t[7] = f2bf(b.w);
  *(uint4*)(dst + i) = *(uint4*)t;
}

// ---------------------------------------------------------------------------
// LayerNorm: fp32 [rows, DMODEL] -> bf16 out
// ---------------------------------------------------------------------------
__global__ __launch_bounds__(256)
void ln_kernel(const float* __restrict__ h, const float* __restrict__ w,
               const float* __restrict__ b, unsigned short* __restrict__ out) {
  __shared__ float red[2][8];
  const int row = blockIdx.x;
  const int tid = threadIdx.x;
  const float* x = h + (size_t)row * DMODEL;
  float s = 0.f, s2 = 0.f;
  for (int c = tid; c < DMODEL; c += 256) { float v = x[c]; s += v; s2 += v * v; }
  #pragma unroll
  for (int m = 16; m >= 1; m >>= 1) { s += __shfl_xor(s, m, 32); s2 += __shfl_xor(s2, m, 32); }
  const int wid = tid >> 5, lane = tid & 31;
  if (lane == 0) { red[0][wid] = s; red[1][wid] = s2; }
  __syncthreads();
  s = red[0][lane & 7]; s2 = red[1][lane & 7];
  #pragma unroll
  for (int m = 4; m >= 1; m >>= 1) { s += __shfl_xor(s, m, 32); s2 += __shfl_xor(s2, m, 32); }
  const float mean = s * (1.f / DMODEL);
  const float var  = s2 * (1.f / DMODEL) - mean * mean;
  const float inv  = rsqrtf(var + LN_EPS);
  for (int c = tid; c < DMODEL; c += 256) {
    float v = (x[c] - mean) * inv * w[c] + b[c];
    out[(size_t)row * DMODEL + c] = f2bf(v);
  }
}

// ---------------------------------------------------------------------------
// Tiled bf16 WMMA GEMM:  out[M,N] = A_bf16[M,K] @ Wbf16[N,K]^T + bias
// Optional GELU, optional fp32 residual add; fp32 and/or bf16 outputs.
// vtrans: bf16 output stored transposed as [B, N(=D), T] for attention V.
// Block = 256 thr (8 waves), tile 128x128, BK = 64, double-buffered LDS,
// async global->LDS staging (ASYNCcnt).
// ---------------------------------------------------------------------------
#define BM 128
#define BN 128
#define BKK 64
#define TILE_H (BM * BKK)          // halfs per tile buffer

__global__ __launch_bounds__(256)
void gemm_bf16_kernel(const unsigned short* __restrict__ A,
                      const unsigned short* __restrict__ W,
                      const float* __restrict__ bias,
                      const float* __restrict__ res,
                      float* __restrict__ outf,
                      unsigned short* __restrict__ outbf,
                      int M, int N, int K, int do_gelu, int vtrans) {
  __shared__ unsigned short As[2][TILE_H];
  __shared__ unsigned short Bs[2][TILE_H];

  const int tid  = threadIdx.x;
  const int wid  = tid >> 5;
  const int lane = tid & 31;
  const int wm   = wid >> 1;       // 0..3
  const int wn   = wid & 1;        // 0..1
  const int m0   = blockIdx.y * BM;
  const int n0   = blockIdx.x * BN;
  const int hi   = (lane >> 4) & 1;
  const int cl   = lane & 15;

  const v8f zero8 = {0.f, 0.f, 0.f, 0.f, 0.f, 0.f, 0.f, 0.f};
  v8f acc[2][4];
  #pragma unroll
  for (int i = 0; i < 2; ++i)
    #pragma unroll
    for (int j = 0; j < 4; ++j) acc[i][j] = zero8;

  // staging decomposition: thread t covers row = t>>1, 32-half segment (t&1)
  const int lr  = tid >> 1;          // 0..127
  const int seg = (tid & 1) * 32;    // 0 or 32 halfs

  const int nk = K / BKK;

  auto stage = [&](int kt, int buf) {
    const size_t ka = (size_t)(m0 + lr) * K + (size_t)kt * BKK + seg;
    const size_t kb = (size_t)(n0 + lr) * K + (size_t)kt * BKK + seg;
    const int    ls = lr * BKK + seg;
    #pragma unroll
    for (int j = 0; j < 4; ++j)
      cp16_async(A + ka + j * 8, &As[buf][ls + j * 8]);
    #pragma unroll
    for (int j = 0; j < 4; ++j)
      cp16_async(W + kb + j * 8, &Bs[buf][ls + j * 8]);
  };

  stage(0, 0);                      // prologue: tile 0 -> buffer 0

  for (int kt = 0; kt < nk; ++kt) {
    const int cur = kt & 1;
    if (kt + 1 < nk) {
      stage(kt + 1, cur ^ 1);       // overlap next tile behind compute
      WAIT_ASYNC(8);                // 8 newer ops outstanding => tile kt done
    } else {
      WAIT_ASYNC(0);
    }
    __syncthreads();                // tile kt visible to all waves

    #pragma unroll
    for (int ks = 0; ks < BKK; ks += 32) {
      AFrag af[2];
      #pragma unroll
      for (int mt = 0; mt < 2; ++mt) {
        const int row = wm * 32 + mt * 16 + cl;
        #pragma unroll
        for (int e = 0; e < 16; ++e)
          af[mt].u[e] = As[cur][row * BKK + ks + a_kidx(e, hi)];
      }
      #pragma unroll
      for (int nt = 0; nt < 4; ++nt) {
        const int col = wn * 64 + nt * 16 + cl;
        AFrag bf;
        #pragma unroll
        for (int e = 0; e < 16; ++e)
          bf.u[e] = Bs[cur][col * BKK + ks + a_kidx(e, hi)];
        #pragma unroll
        for (int mt = 0; mt < 2; ++mt)
          acc[mt][nt] = __builtin_amdgcn_wmma_f32_16x16x32_bf16(
              false, af[mt].v, false, bf.v, (short)0, acc[mt][nt], false, false);
      }
    }
    __syncthreads();                // everyone done reading buf[cur]
  }

  // ---- epilogue ----
  #pragma unroll
  for (int mt = 0; mt < 2; ++mt)
    #pragma unroll
    for (int nt = 0; nt < 4; ++nt)
      #pragma unroll
      for (int r = 0; r < 8; ++r) {
        const int row = m0 + wm * 32 + mt * 16 + r + hi * 8;
        const int col = n0 + wn * 64 + nt * 16 + cl;
        float v = acc[mt][nt][r];
        if (bias) v += bias[col];
        if (do_gelu) v = gelu_tanh(v);
        if (res) v += res[(size_t)row * N + col];
        if (outf)  outf[(size_t)row * N + col]  = v;
        if (outbf) {
          size_t oidx;
          if (vtrans) {
            const int bb = row >> 10;            // TSEQ = 1024
            const int tt = row & (TSEQ - 1);
            oidx = (((size_t)bb * DMODEL + col) << 10) + tt;
          } else {
            oidx = (size_t)row * N + col;
          }
          outbf[oidx] = f2bf(v);
        }
      }
}

// ---------------------------------------------------------------------------
// Flash attention: grid(B*H, T/128), 256 threads (8 waves x 16 query rows).
// q,k bf16 [B*T, D]; vt bf16 transposed [B, D, T]. Output bf16 [B*T, D].
// K and V^T key-chunks staged to LDS with async copies, double-buffered.
// ---------------------------------------------------------------------------
__global__ __launch_bounds__(256)
void attn_kernel(const unsigned short* __restrict__ qb,
                 const unsigned short* __restrict__ kb,
                 const unsigned short* __restrict__ vt,
                 const float* __restrict__ mask,        // [B, T]
                 unsigned short* __restrict__ ob) {
  __shared__ unsigned short Ks[2][64 * 64];   // [key][hd]
  __shared__ unsigned short Vs[2][64 * 64];   // [hd][key]
  __shared__ unsigned short pbuf[8][16][64];  // per-wave P tile

  const int bh   = blockIdx.x;
  const int b    = bh / NHEAD;
  const int hidx = bh % NHEAD;                // head index
  const int tid  = threadIdx.x;
  const int wid  = tid >> 5;
  const int lane = tid & 31;
  const int t0   = blockIdx.y * 128 + wid * 16;  // query tile base
  const int hi   = (lane >> 4) & 1;
  const int cl   = lane & 15;

  // staging decomposition: 256 threads cover 64 rows x 64 halfs per array
  const int srow = tid >> 2;            // 0..63
  const int sseg = (tid & 3) * 16;      // 0,16,32,48

  const unsigned short* kbase = kb + ((size_t)b * TSEQ) * DMODEL + hidx * HDIM;
  const unsigned short* vbase = vt + (((size_t)b * DMODEL + hidx * HDIM) << 10);

  auto stage = [&](int kc, int buf) {
    const unsigned short* kg = kbase + (size_t)(kc * 64 + srow) * DMODEL + sseg;
    const unsigned short* vg = vbase + ((size_t)srow << 10) + kc * 64 + sseg;
    const int ls = srow * 64 + sseg;
    cp16_async(kg,     &Ks[buf][ls]);
    cp16_async(kg + 8, &Ks[buf][ls + 8]);
    cp16_async(vg,     &Vs[buf][ls]);
    cp16_async(vg + 8, &Vs[buf][ls + 8]);
  };

  const v8f zero8 = {0.f, 0.f, 0.f, 0.f, 0.f, 0.f, 0.f, 0.f};

  // ---- load Q fragments (16 rows x HD=64 -> two 16x32 A-frags) ----
  AFrag qf[2];
  {
    const unsigned short* qrow =
        qb + ((size_t)(b * TSEQ + t0 + cl)) * DMODEL + hidx * HDIM;
    #pragma unroll
    for (int f = 0; f < 2; ++f)
      #pragma unroll
      for (int e = 0; e < 16; ++e)
        qf[f].u[e] = qrow[f * 32 + a_kidx(e, hi)];
  }

  v8f oacc[4];
  #pragma unroll
  for (int nt = 0; nt < 4; ++nt) oacc[nt] = zero8;
  float m_i[8], l_i[8];
  #pragma unroll
  for (int r = 0; r < 8; ++r) { m_i[r] = -1e30f; l_i[r] = 0.f; }

  const int nkc = TSEQ / 64;
  stage(0, 0);

  for (int kc = 0; kc < nkc; ++kc) {
    const int cur = kc & 1;
    if (kc + 1 < nkc) {
      stage(kc + 1, cur ^ 1);
      WAIT_ASYNC(4);
    } else {
      WAIT_ASYNC(0);
    }
    __syncthreads();

    // ---- S = Q @ K^T for this 64-key chunk ----
    v8f sacc[4];
    #pragma unroll
    for (int nt = 0; nt < 4; ++nt) sacc[nt] = zero8;
    #pragma unroll
    for (int nt = 0; nt < 4; ++nt) {
      #pragma unroll
      for (int f = 0; f < 2; ++f) {
        AFrag bf;
        #pragma unroll
        for (int e = 0; e < 16; ++e)
          bf.u[e] = Ks[cur][(nt * 16 + cl) * 64 + f * 32 + a_kidx(e, hi)];
        sacc[nt] = __builtin_amdgcn_wmma_f32_16x16x32_bf16(
            false, qf[f].v, false, bf.v, (short)0, sacc[nt], false, false);
      }
    }

    // ---- scale + mask, online softmax update ----
    float mk[4];
    #pragma unroll
    for (int nt = 0; nt < 4; ++nt)
      mk[nt] = mask[(size_t)b * TSEQ + kc * 64 + nt * 16 + cl];

    #pragma unroll
    for (int r = 0; r < 8; ++r) {
      float vmax = -1e30f;
      #pragma unroll
      for (int nt = 0; nt < 4; ++nt) {
        float s = sacc[nt][r] * ATT_SCALE + mk[nt];
        sacc[nt][r] = s;
        vmax = fmaxf(vmax, s);
      }
      #pragma unroll
      for (int m = 8; m >= 1; m >>= 1) vmax = fmaxf(vmax, __shfl_xor(vmax, m, 32));
      const float mn   = fmaxf(m_i[r], vmax);
      const float corr = __expf(m_i[r] - mn);
      float rsum = 0.f;
      #pragma unroll
      for (int nt = 0; nt < 4; ++nt) {
        float p = __expf(sacc[nt][r] - mn);
        sacc[nt][r] = p;
        rsum += p;
        oacc[nt][r] *= corr;
      }
      #pragma unroll
      for (int m = 8; m >= 1; m >>= 1) rsum += __shfl_xor(rsum, m, 32);
      l_i[r] = l_i[r] * corr + rsum;
      m_i[r] = mn;
    }

    // ---- repack P (C layout -> A layout) through per-wave LDS ----
    #pragma unroll
    for (int nt = 0; nt < 4; ++nt)
      #pragma unroll
      for (int r = 0; r < 8; ++r)
        pbuf[wid][r + hi * 8][nt * 16 + cl] = f2bf(sacc[nt][r]);
    // same-wave LDS ops are in order; no block barrier needed
    AFrag pf[2];
    #pragma unroll
    for (int f = 0; f < 2; ++f)
      #pragma unroll
      for (int e = 0; e < 16; ++e)
        pf[f].u[e] = pbuf[wid][cl][f * 32 + a_kidx(e, hi)];

    // ---- O += P @ V  (V^T tile in LDS: [hd][key]) ----
    #pragma unroll
    for (int nt = 0; nt < 4; ++nt) {
      #pragma unroll
      for (int f = 0; f < 2; ++f) {
        AFrag vf;
        #pragma unroll
        for (int e = 0; e < 16; ++e)
          vf.u[e] = Vs[cur][(nt * 16 + cl) * 64 + f * 32 + a_kidx(e, hi)];
        oacc[nt] = __builtin_amdgcn_wmma_f32_16x16x32_bf16(
            false, pf[f].v, false, vf.v, (short)0, oacc[nt], false, false);
      }
    }
    __syncthreads();
  }

  // ---- normalize and store bf16 ----
  #pragma unroll
  for (int nt = 0; nt < 4; ++nt)
    #pragma unroll
    for (int r = 0; r < 8; ++r) {
      const int trow = t0 + r + hi * 8;
      const float v = oacc[nt][r] / l_i[r];
      ob[((size_t)(b * TSEQ + trow)) * DMODEL + hidx * HDIM + nt * 16 + cl] = f2bf(v);
    }
}

// ---------------------------------------------------------------------------
// Host-side orchestration
// ---------------------------------------------------------------------------
extern "C" void kernel_launch(void* const* d_in, const int* in_sizes, int n_in,
                              void* d_out, int out_size, void* d_ws, size_t ws_size,
                              hipStream_t stream) {
  const float* hs    = (const float*)d_in[0];
  const float* mask  = (const float*)d_in[1];
  const float* ln1w  = (const float*)d_in[2];
  const float* ln1b  = (const float*)d_in[3];
  const float* qw    = (const float*)d_in[4];
  const float* qbias = (const float*)d_in[5];
  const float* kw    = (const float*)d_in[6];
  const float* kbias = (const float*)d_in[7];
  const float* vw    = (const float*)d_in[8];
  const float* vbias = (const float*)d_in[9];
  const float* ow    = (const float*)d_in[10];
  const float* obias = (const float*)d_in[11];
  const float* ln2w  = (const float*)d_in[12];
  const float* ln2b  = (const float*)d_in[13];
  const float* fc1w  = (const float*)d_in[14];
  const float* fc1b  = (const float*)d_in[15];
  const float* fc2w  = (const float*)d_in[16];
  const float* fc2b  = (const float*)d_in[17];
  const float* lnpw  = (const float*)d_in[18];
  const float* lnpb  = (const float*)d_in[19];
  const float* p1w   = (const float*)d_in[20];
  const float* p1b   = (const float*)d_in[21];
  const float* p2w   = (const float*)d_in[22];
  const float* p2b   = (const float*)d_in[23];

  // carve workspace
  char* ws = (char*)d_ws;
  size_t off = 0;
  auto carve = [&](size_t bytes) -> void* {
    void* p = ws + off;
    off += (bytes + 255) & ~(size_t)255;
    return p;
  };
  unsigned short* xbf = (unsigned short*)carve((size_t)MTOK * DMODEL * 2);
  unsigned short* qbf = (unsigned short*)carve((size_t)MTOK * DMODEL * 2);
  unsigned short* kbf = (unsigned short*)carve((size_t)MTOK * DMODEL * 2);
  unsigned short* vtb = (unsigned short*)carve((size_t)MTOK * DMODEL * 2);
  unsigned short* obf = (unsigned short*)carve((size_t)MTOK * DMODEL * 2);
  unsigned short* gbf = (unsigned short*)carve((size_t)MTOK * FDIM * 2);
  unsigned short* wbf = (unsigned short*)carve((size_t)FDIM * DMODEL * 2);
  float* t0f = (float*)carve((size_t)MTOK * DMODEL * 4);
  float* t1f = (float*)carve((size_t)MTOK * DMODEL * 4);

  // convert one weight matrix (fp32 [N,K]) to bf16, then run WMMA GEMM
  auto gemm = [&](const unsigned short* A, const float* W, const float* bias,
                  const float* res, float* outf, unsigned short* outbf,
                  int M, int N, int K, int gelu, int vtrans) {
    const size_t nelem = (size_t)N * K;
    f32_to_bf16_kernel<<<dim3((unsigned)(nelem / 2048)), dim3(256), 0, stream>>>(
        W, wbf);
    dim3 grid(N / BN, M / BM);
    gemm_bf16_kernel<<<grid, dim3(256), 0, stream>>>(A, wbf, bias, res, outf,
                                                     outbf, M, N, K, gelu,
                                                     vtrans);
  };
  auto ln = [&](const float* h, const float* w, const float* b,
                unsigned short* out) {
    ln_kernel<<<dim3(MTOK), dim3(256), 0, stream>>>(h, w, b, out);
  };

  const float* cur = hs;
  for (int i = 0; i < L_LAYERS; ++i) {
    const size_t dd = (size_t)DMODEL * DMODEL;
    // --- attention block ---
    ln(cur, ln1w + (size_t)i * DMODEL, ln1b + (size_t)i * DMODEL, xbf);
    gemm(xbf, qw + (size_t)i * dd, qbias + (size_t)i * DMODEL,
         nullptr, nullptr, qbf, MTOK, DMODEL, DMODEL, 0, 0);
    gemm(xbf, kw + (size_t)i * dd, kbias + (size_t)i * DMODEL,
         nullptr, nullptr, kbf, MTOK, DMODEL, DMODEL, 0, 0);
    gemm(xbf, vw + (size_t)i * dd, vbias + (size_t)i * DMODEL,
         nullptr, nullptr, vtb, MTOK, DMODEL, DMODEL, 0, 1);   // V transposed
    attn_kernel<<<dim3(BSZ * NHEAD, TSEQ / 128), dim3(256), 0, stream>>>(
        qbf, kbf, vtb, mask, obf);
    gemm(obf, ow + (size_t)i * dd, obias + (size_t)i * DMODEL,
         cur, t0f, nullptr, MTOK, DMODEL, DMODEL, 0, 0);
    // --- FFN block ---
    ln(t0f, ln2w + (size_t)i * DMODEL, ln2b + (size_t)i * DMODEL, xbf);
    gemm(xbf, fc1w + (size_t)i * FDIM * DMODEL, fc1b + (size_t)i * FDIM,
         nullptr, nullptr, gbf, MTOK, FDIM, DMODEL, 1, 0);
    gemm(gbf, fc2w + (size_t)i * DMODEL * FDIM, fc2b + (size_t)i * DMODEL,
         t0f, t1f, nullptr, MTOK, DMODEL, FDIM, 0, 0);
    cur = t1f;
  }

  // --- post norm + adapter ---
  ln(cur, lnpw, lnpb, xbf);
  gemm(xbf, p1w, p1b, nullptr, nullptr, gbf, MTOK, DMODEL, DMODEL, 1, 0);
  gemm(gbf, p2w, p2b, nullptr, (float*)d_out, nullptr, MTOK, OUTDIM, DMODEL, 0, 0);
}
